// STCNModel_24696061952617
// MI455X (gfx1250) — matrix operand order repaired and checked by reading, not compiled, not run on hardware
//
#include <hip/hip_runtime.h>

// ---------------------------------------------------------------------------
// STCN read_memory for MI455X (gfx1250, wave32, WMMA).
//
// Shapes: B=4, CK=64, T=16, H=W=30 -> HW=900, THW=14400; N=8, CV=512.
// FLOPs ~113G (AV GEMM dominates: 8 * 512*900*14400*2 = 106 GFLOP).
// Strategy: bf16 WMMA (v_wmma_f32_16x16x32_bf16) with f32 accumulation.
// bf16 copies of V (118MB) + in-place bf16 P rows keep the hot working set
// close to the 192MB L2; softmax rows (57.6KB) are staged whole in LDS.
// AV GEMM uses 4-wave workgroups: B panel double-buffered in LDS via
// GLOBAL_LOAD_ASYNC_TO_LDS_B128 (ASYNCcnt), each wave 4 m-tiles x 2 q-tiles.
// ---------------------------------------------------------------------------

#define B_   4
#define CK_  64
#define T_   16
#define H_   30
#define W_   30
#define HW_  (H_ * W_)       // 900
#define THW_ (T_ * H_ * W_)  // 14400
#define N_   8
#define CV_  512
#define QP_  928             // HW padded to 58 q-tiles (even, for q-pairs)
#define QTILES_ (QP_ / 16)   // 58
#define NK_  (THW_ / 32)     // 450 K-steps of 32

typedef __attribute__((ext_vector_type(16))) __bf16 v16bf;
typedef __attribute__((ext_vector_type(8)))  __bf16 v8bf;
typedef __attribute__((ext_vector_type(8)))  float  v8f;
typedef int v4i_ __attribute__((vector_size(4 * sizeof(int))));

union V16 { v16bf v; v8bf h[2]; };

#if defined(__has_builtin)
#if __has_builtin(__builtin_amdgcn_global_load_async_to_lds_b128)
#define HAVE_ASYNC_LDS 1
#endif
#endif
#ifndef HAVE_ASYNC_LDS
#define HAVE_ASYNC_LDS 0
#endif

#if HAVE_ASYNC_LDS
// Exact param types per compiler diagnostic: AS(1)/AS(3) pointers to int4.
#define ASYNC_GPTR(x) ((__attribute__((address_space(1))) v4i_*)(x))
#define ASYNC_LPTR(x) ((__attribute__((address_space(3))) v4i_*)(x))
#endif

__device__ __forceinline__ void wait_asynccnt0() {
#if defined(__has_builtin) && __has_builtin(__builtin_amdgcn_s_wait_asynccnt)
  __builtin_amdgcn_s_wait_asynccnt(0);
#else
  asm volatile("s_wait_asynccnt 0x0" ::: "memory");
#endif
}

// ---------------------------------------------------------------------------
// Prep kernels: fp32 -> bf16 conversions into WMMA-friendly layouts.
// ---------------------------------------------------------------------------

// mem_keys [B,CK,THW] -> mkT bf16 [B,THW,CK]; a_sq[b,t] = sum_c mk^2 (f32).
__global__ void k_prep_keys(const float* __restrict__ mk,
                            __bf16* __restrict__ mkT,
                            float* __restrict__ a_sq) {
  int idx = blockIdx.x * blockDim.x + threadIdx.x;  // b*THW + t
  if (idx >= B_ * THW_) return;
  int b = idx / THW_;
  int t = idx % THW_;
  const float* src = mk + (size_t)b * CK_ * THW_ + t;
  __bf16* dst = mkT + (size_t)idx * CK_;
  float s = 0.f;
#pragma unroll 8
  for (int c = 0; c < CK_; ++c) {
    float v = src[(size_t)c * THW_];
    s += v * v;
    dst[c] = (__bf16)v;
  }
  a_sq[idx] = s;
}

// qk [B,CK,HW] -> qT bf16 [B,QP,CK]; rows q>=900 zero-padded.
__global__ void k_prep_q(const float* __restrict__ qk,
                         __bf16* __restrict__ qT) {
  int idx = blockIdx.x * blockDim.x + threadIdx.x;  // b*QP + q
  if (idx >= B_ * QP_) return;
  int b = idx / QP_;
  int q = idx % QP_;
  __bf16* dst = qT + (size_t)idx * CK_;
  if (q < HW_) {
    const float* src = qk + (size_t)b * CK_ * HW_ + q;
#pragma unroll 8
    for (int c = 0; c < CK_; ++c) dst[c] = (__bf16)src[(size_t)c * HW_];
  } else {
#pragma unroll 8
    for (int c = 0; c < CK_; ++c) dst[c] = (__bf16)0.f;
  }
}

// mem_values [N,CV,THW] fp32 -> bf16, same layout.
__global__ void k_cvt_values(const float* __restrict__ mv,
                             __bf16* __restrict__ moB) {
  size_t i = ((size_t)blockIdx.x * blockDim.x + threadIdx.x) * 4;
  const size_t n = (size_t)N_ * CV_ * THW_;
  if (i + 3 < n) {
    float4 f = *(const float4*)(mv + i);
    moB[i + 0] = (__bf16)f.x;
    moB[i + 1] = (__bf16)f.y;
    moB[i + 2] = (__bf16)f.z;
    moB[i + 3] = (__bf16)f.w;
  } else {
    for (; i < n; ++i) moB[i] = (__bf16)mv[i];
  }
}

// ---------------------------------------------------------------------------
// QK GEMM: logits[b][q][t] = (2 * sum_c mk[b,c,t]*q[b,c,q] - a_sq[b,t]) / 8.
// One wave per 16x16 (t,q) tile; K = 64 -> two 16x16x32 bf16 WMMAs.
// ---------------------------------------------------------------------------
__global__ void __launch_bounds__(32)
k_qk_gemm(const __bf16* __restrict__ mkT,
          const __bf16* __restrict__ qT,
          const float* __restrict__ a_sq,
          float* __restrict__ logits) {
  const int tt   = blockIdx.x;        // t tile, 0..899
  const int qt   = blockIdx.y;        // q tile, 0..57
  const int b    = blockIdx.z;
  const int lane = threadIdx.x;
  const int g    = lane >> 4;
  const int lm   = lane & 15;

  const __bf16* Arow = mkT + ((size_t)b * THW_ + (size_t)tt * 16 + lm) * CK_;
  const __bf16* Brow = qT + ((size_t)b * QP_ + (size_t)qt * 16 + lm) * CK_;

  v8f acc = {};
#pragma unroll
  for (int kk = 0; kk < 2; ++kk) {
    V16 A;
    A.h[0] = *(const v8bf*)(Arow + kk * 32 + g * 16);
    A.h[1] = *(const v8bf*)(Arow + kk * 32 + g * 16 + 8);
    v16bf Bv = *(const v16bf*)(Brow + kk * 32 + g * 16);
    acc = __builtin_amdgcn_wmma_f32_16x16x32_bf16(
        false, A.v, false, Bv, (short)0, acc, false, false);
  }

  const int t0 = tt * 16 + g * 8;     // D: VGPR r -> M = r + g*8
  const float* asq = a_sq + (size_t)b * THW_ + t0;
  float* out = logits + ((size_t)b * QP_ + (size_t)qt * 16 + lm) * THW_ + t0;
#pragma unroll
  for (int r = 0; r < 8; ++r)
    out[r] = (2.f * acc[r] - asq[r]) * 0.125f;  // 1/sqrt(64)
}

// ---------------------------------------------------------------------------
// Softmax over THW per (b,q) row; full 57.6KB row staged in LDS (320KB/WGP),
// bf16 probabilities written IN PLACE over the f32 row.
// ---------------------------------------------------------------------------
__device__ __forceinline__ float wave_max_f(float v) {
#pragma unroll
  for (int off = 16; off > 0; off >>= 1) v = fmaxf(v, __shfl_xor(v, off, 32));
  return v;
}
__device__ __forceinline__ float wave_sum_f(float v) {
#pragma unroll
  for (int off = 16; off > 0; off >>= 1) v += __shfl_xor(v, off, 32);
  return v;
}

__global__ void __launch_bounds__(256)
k_softmax(float* __restrict__ logits) {
  const int row = blockIdx.x;            // b*QP + q
  const int q   = row % QP_;
  float* xr = logits + (size_t)row * THW_;
  __bf16* yr = (__bf16*)xr;
  const int tid = threadIdx.x;

  if (q >= HW_) {                        // padded row: uniform branch
    for (int i = tid; i < THW_; i += 256) yr[i] = (__bf16)0.f;
    return;
  }

  __shared__ float red[8];
  extern __shared__ float rowbuf[];      // THW_ floats

  float m = -3.4e38f;
  for (int i = tid; i < THW_; i += 256) {
    float v = xr[i];
    rowbuf[i] = v;
    m = fmaxf(m, v);
  }
  m = wave_max_f(m);
  if ((tid & 31) == 0) red[tid >> 5] = m;
  __syncthreads();
  float mrow = red[0];
#pragma unroll
  for (int i = 1; i < 8; ++i) mrow = fmaxf(mrow, red[i]);
  __syncthreads();

  float s = 0.f;
  for (int i = tid; i < THW_; i += 256) s += __expf(rowbuf[i] - mrow);
  s = wave_sum_f(s);
  if ((tid & 31) == 0) red[tid >> 5] = s;
  __syncthreads();
  float srow = 0.f;
#pragma unroll
  for (int i = 0; i < 8; ++i) srow += red[i];
  const float rinv = __frcp_rn(srow);
  __syncthreads();

  for (int i = tid; i < THW_; i += 256)
    yr[i] = (__bf16)(__expf(rowbuf[i] - mrow) * rinv);
}

// ---------------------------------------------------------------------------
// AV GEMM: out[n][c][q] = sum_t V[n,c,t] * P[bmap[n],q,t].
// 4-wave workgroup computes a 256x32 output block: wave w owns m-tiles
// [mg*256 + w*64, +64) x 2 q-tiles (8 accumulators, 8 WMMAs per K-step).
// The shared 2KB B panel (2 q-tiles x 16 rows x 32 K) is double-buffered in
// LDS via GLOBAL_LOAD_ASYNC_TO_LDS_B128: each of the 128 threads copies one
// 16B chunk for the next K-step, s_wait_asynccnt + barrier at loop top.
// ---------------------------------------------------------------------------
__global__ void __launch_bounds__(128)
k_av_gemm(const __bf16* __restrict__ moB,
          const char* __restrict__ Pbytes,     // bf16 P rows, stride 4*THW B
          const int* __restrict__ bmap,
          float* __restrict__ out) {
  const int mg   = blockIdx.x;        // 0..1 : c block [mg*256, +256)
  const int qp   = blockIdx.y;        // 0..28 : q tiles qp*2, qp*2+1
  const int n    = blockIdx.z;
  const int b    = bmap[n];
  const int tid  = threadIdx.x;
  const int wave = tid >> 5;
  const int lane = tid & 31;
  const int g    = lane >> 4;
  const int lm   = lane & 15;

#if HAVE_ASYNC_LDS
  __shared__ __attribute__((aligned(32))) __bf16 Bs[2][2][16][32];
  // Per-thread async-copy slot: 128 threads x 16B = 2KB = full panel.
  const int ctile  = tid >> 6;         // which q-tile
  const int crow   = (tid >> 2) & 15;  // q row within tile
  const int cchunk = tid & 3;          // 16B chunk within the 64B K-row
  const char* gsrc = Pbytes +
      ((size_t)b * QP_ + (size_t)(qp * 2 + ctile) * 16 + crow) *
          (size_t)THW_ * 4 +
      (size_t)cchunk * 16;
#else
  const __bf16* Brow0 = (const __bf16*)(Pbytes +
      ((size_t)b * QP_ + (size_t)(qp * 2) * 16 + lm) * (size_t)THW_ * 4);
  const __bf16* Brow1 = (const __bf16*)(Pbytes +
      ((size_t)b * QP_ + (size_t)(qp * 2 + 1) * 16 + lm) * (size_t)THW_ * 4);
#endif

  const __bf16* Abase =
      moB + ((size_t)n * CV_ + (size_t)mg * 256 + wave * 64 + lm) * THW_;

  v8f acc[4][2] = {{{}, {}}, {{}, {}}, {{}, {}}, {{}, {}}};

#if HAVE_ASYNC_LDS
  // Prologue: stage K-step 0 into buffer 0.
  __builtin_amdgcn_global_load_async_to_lds_b128(
      ASYNC_GPTR(gsrc), ASYNC_LPTR(&Bs[0][ctile][crow][cchunk * 8]), 0, 0);
#endif

  for (int it = 0; it < NK_; ++it) {
    const int k0 = it * 32;
#if HAVE_ASYNC_LDS
    const int p = it & 1;
    wait_asynccnt0();                  // my async copies for buf p landed
    __syncthreads();                   // everyone's landed; prev reads done
    if (it + 1 < NK_) {
      __builtin_amdgcn_global_load_async_to_lds_b128(
          ASYNC_GPTR(gsrc + (size_t)(k0 + 32) * 2),
          ASYNC_LPTR(&Bs[p ^ 1][ctile][crow][cchunk * 8]), 0, 0);
    }
    v16bf Bv0 = *(const v16bf*)&Bs[p][0][lm][g * 16];
    v16bf Bv1 = *(const v16bf*)&Bs[p][1][lm][g * 16];
#else
    v16bf Bv0 = *(const v16bf*)(Brow0 + k0 + g * 16);
    v16bf Bv1 = *(const v16bf*)(Brow1 + k0 + g * 16);
#endif
#pragma unroll
    for (int m = 0; m < 4; ++m) {
      const __bf16* Ar = Abase + (size_t)m * 16 * THW_;
      if (k0 + 512 < THW_)             // stream-ahead -> global_prefetch_b8
        __builtin_prefetch(Ar + k0 + 512 + g * 8, 0, 1);
      V16 A;
      A.h[0] = *(const v8bf*)(Ar + k0 + g * 16);
      A.h[1] = *(const v8bf*)(Ar + k0 + g * 16 + 8);
      acc[m][0] = __builtin_amdgcn_wmma_f32_16x16x32_bf16(
          false, A.v, false, Bv0, (short)0, acc[m][0], false, false);
      acc[m][1] = __builtin_amdgcn_wmma_f32_16x16x32_bf16(
          false, A.v, false, Bv1, (short)0, acc[m][1], false, false);
    }
  }

#pragma unroll
  for (int qi = 0; qi < 2; ++qi) {
    const int q = (qp * 2 + qi) * 16 + lm;
    if (q < HW_) {
#pragma unroll
      for (int m = 0; m < 4; ++m) {
        const int c0 = mg * 256 + wave * 64 + m * 16 + g * 8;  // M = r + g*8
        float* o = out + ((size_t)n * CV_ + c0) * HW_ + q;
#pragma unroll
        for (int r = 0; r < 8; ++r) o[(size_t)r * HW_] = acc[m][qi][r];
      }
    }
  }
}

// ---------------------------------------------------------------------------
// Launch. Inputs: qk, mem_keys, mem_values, broadcast_map. Output f32
// [N,CV,H,W]. Workspace (~340MB): mkT 7.4MB | a_sq 0.23MB | qT 0.48MB |
// logits f32 [B,QP,THW] 213.8MB (bf16 P overwrites rows in place) |
// moB bf16 118MB.
// ---------------------------------------------------------------------------
extern "C" void kernel_launch(void* const* d_in, const int* in_sizes, int n_in,
                              void* d_out, int out_size, void* d_ws,
                              size_t ws_size, hipStream_t stream) {
  const float* qk   = (const float*)d_in[0];
  const float* mk   = (const float*)d_in[1];
  const float* mv   = (const float*)d_in[2];
  const int*   bmap = (const int*)d_in[3];
  float* out = (float*)d_out;

  char* ws = (char*)d_ws;
  size_t off = 0;
  auto alloc = [&](size_t bytes) -> size_t {
    off = (off + 255) & ~(size_t)255;
    size_t o = off;
    off += bytes;
    return o;
  };
  const size_t o_mkT = alloc((size_t)B_ * THW_ * CK_ * 2);
  const size_t o_asq = alloc((size_t)B_ * THW_ * 4);
  const size_t o_qT  = alloc((size_t)B_ * QP_ * CK_ * 2);
  const size_t o_lg  = alloc((size_t)B_ * QP_ * THW_ * 4);
  const size_t o_moB = alloc((size_t)N_ * CV_ * THW_ * 2);
  if (off > ws_size) return;  // workspace too small; cannot report from here

  __bf16* mkT  = (__bf16*)(ws + o_mkT);
  float*  a_sq = (float*)(ws + o_asq);
  __bf16* qT   = (__bf16*)(ws + o_qT);
  float*  lg   = (float*)(ws + o_lg);
  __bf16* moB  = (__bf16*)(ws + o_moB);

  {
    int n = B_ * THW_;
    k_prep_keys<<<(n + 255) / 256, 256, 0, stream>>>(mk, mkT, a_sq);
  }
  {
    int n = B_ * QP_;
    k_prep_q<<<(n + 255) / 256, 256, 0, stream>>>(qk, qT);
  }
  {
    size_t n4 = ((size_t)N_ * CV_ * THW_ + 3) / 4;
    k_cvt_values<<<(unsigned)((n4 + 255) / 256), 256, 0, stream>>>(mv, moB);
  }
  {
    dim3 grid(THW_ / 16, QTILES_, B_);
    k_qk_gemm<<<grid, 32, 0, stream>>>(mkT, qT, a_sq, lg);
  }
  {
    dim3 grid(B_ * QP_);
    k_softmax<<<grid, 256, THW_ * sizeof(float), stream>>>(lg);
  }
  {
    dim3 grid(CV_ / 256, QTILES_ / 2, N_);
    k_av_gemm<<<grid, 128, 0, stream>>>(moB, (const char*)lg, bmap, out);
  }
}
